// SubgraphGraphormer_53575422050475
// MI455X (gfx1250) — compile-verified, matrix operand order checked by reading, not run on hardware
//
#include <hip/hip_runtime.h>
#include <hip/hip_bf16.h>
#include <math.h>

#define L_ 4
#define B_ 16
#define S_ 512
#define H_ 768
#define NH_ 12
#define FF_ 3072
#define D_ 64
#define M_ (B_*S_)   /* 8192 tokens */

// Async global->LDS path (gfx1250). Preprocessor-gated so the host pass (which
// doesn't know the builtin) takes the synchronous fallback consistently.
#if defined(__HIP_DEVICE_COMPILE__) && __has_builtin(__builtin_amdgcn_global_load_async_to_lds_b128)
#define ASYNC_LDS 1
#else
#define ASYNC_LDS 0
#endif

typedef __attribute__((ext_vector_type(16))) __bf16 v16bf;
typedef __attribute__((ext_vector_type(8)))  float  v8f;
typedef __attribute__((ext_vector_type(4)))  int    v4i;

struct __attribute__((aligned(16))) U128 { unsigned int x[4]; };

__device__ __forceinline__ void wait_async0() {
#if ASYNC_LDS
#if __has_builtin(__builtin_amdgcn_s_wait_asynccnt)
  __builtin_amdgcn_s_wait_asynccnt(0);
#else
  asm volatile("s_wait_asynccnt 0x0" ::: "memory");
#endif
#endif
}

#if ASYNC_LDS
// Copy 64 contiguous bytes global->LDS asynchronously (4 x b128).
// Builtin expects (int4 __device__*, int4 __shared__*, imm offset, imm cpol);
// the instruction offset applies to both global and LDS addresses.
__device__ __forceinline__ void async_cp64(const __bf16* g, __bf16* l) {
  __attribute__((address_space(1))) v4i* gp =
      (__attribute__((address_space(1))) v4i*)g;
  __attribute__((address_space(3))) v4i* lp =
      (__attribute__((address_space(3))) v4i*)l;
  __builtin_amdgcn_global_load_async_to_lds_b128(gp, lp, 0, 0);
  __builtin_amdgcn_global_load_async_to_lds_b128(gp, lp, 16, 0);
  __builtin_amdgcn_global_load_async_to_lds_b128(gp, lp, 32, 0);
  __builtin_amdgcn_global_load_async_to_lds_b128(gp, lp, 48, 0);
}
#endif

// Load one 16x32 bf16 WMMA fragment (A or B/N-major) from LDS.
// Layout (ISA 7.12.2): lane<16 -> row = lane, K = {0..7,16..23};
//                      lane>=16 -> row = lane-16, K = {8..15,24..31}.
__device__ __forceinline__ v16bf ld_frag16(const __bf16* p) {
  union { U128 q[2]; v16bf v; } u;
  u.q[0] = *(const U128*)(p);
  u.q[1] = *(const U128*)(p + 16);
  return u.v;
}

__device__ __forceinline__ v8f wmma_bf16(v16bf a, v16bf b, v8f c) {
  return __builtin_amdgcn_wmma_f32_16x16x32_bf16(false, a, false, b, (short)0, c,
                                                 false, false);
}

// ---------------------------------------------------------------------------
// Weight convert + transpose: src fp32 [K,N] -> dst bf16 [N,K]
// ---------------------------------------------------------------------------
__global__ void wconv_t_kernel(const float* __restrict__ src, __bf16* __restrict__ dst,
                               int K, int N) {
  size_t idx = (size_t)blockIdx.x * blockDim.x + threadIdx.x;
  if (idx >= (size_t)K * N) return;
  int k = (int)(idx / N);
  int n = (int)(idx % N);
  dst[(size_t)n * K + k] = (__bf16)src[idx];
}

__global__ void copy_f32_kernel(const float* __restrict__ s, float* __restrict__ d, int n) {
  int i = blockIdx.x * blockDim.x + threadIdx.x;
  if (i < n) d[i] = s[i];
}

// ---------------------------------------------------------------------------
// LayerNorm over H=768. One block per token, 256 threads (3 elems/thread).
// ---------------------------------------------------------------------------
__global__ void __launch_bounds__(256)
ln_kernel(const float* __restrict__ x, const float* __restrict__ g,
          const float* __restrict__ bt, __bf16* yb, float* yf) {
  const int row = blockIdx.x;
  const float* xr = x + (size_t)row * H_;
  int tid = threadIdx.x;
  float vals[3];
  float s = 0.f, ss = 0.f;
#pragma unroll
  for (int i = 0; i < 3; ++i) {
    float v = xr[tid + i * 256];
    vals[i] = v; s += v; ss += v * v;
  }
#pragma unroll
  for (int o = 16; o > 0; o >>= 1) { s += __shfl_xor(s, o); ss += __shfl_xor(ss, o); }
  __shared__ float rs[8], rss[8];
  int wid = tid >> 5, lane = tid & 31;
  if (lane == 0) { rs[wid] = s; rss[wid] = ss; }
  __syncthreads();
  if (wid == 0) {
    s  = (lane < 8) ? rs[lane]  : 0.f;
    ss = (lane < 8) ? rss[lane] : 0.f;
#pragma unroll
    for (int o = 4; o > 0; o >>= 1) { s += __shfl_xor(s, o); ss += __shfl_xor(ss, o); }
    if (lane == 0) { rs[0] = s; rss[0] = ss; }
  }
  __syncthreads();
  float mean = rs[0] * (1.f / H_);
  float var  = rss[0] * (1.f / H_) - mean * mean;
  float inv  = rsqrtf(var + 1e-5f);
#pragma unroll
  for (int i = 0; i < 3; ++i) {
    int c = tid + i * 256;
    float yv = (vals[i] - mean) * inv * g[c] + bt[c];
    if (yb) yb[(size_t)row * H_ + c] = (__bf16)yv;
    if (yf) yf[(size_t)row * H_ + c] = yv;
  }
}

// ---------------------------------------------------------------------------
// bf16 WMMA GEMM:  C[M,N] = A[M,K] * Bt[N,K]^T + bias (+GELU) (+res)
// 128 threads = 4 waves, block tile 128x128, wave tile 64x64, K-step 32,
// double-buffered LDS with async global->LDS copies.
// ---------------------------------------------------------------------------
template <bool GELU>
__global__ void __launch_bounds__(128)
gemm_bf16_kernel(const __bf16* __restrict__ A, const __bf16* __restrict__ Bt,
                 const float* __restrict__ bias, const float* res,
                 float* outf, __bf16* outb, int M, int N, int K) {
  constexpr int LD = 40;               // padded LDS row stride (bf16 elems)
  __shared__ __bf16 As[2][128 * LD];
  __shared__ __bf16 Bs[2][128 * LD];

  const int m0 = blockIdx.y * 128;
  const int n0 = blockIdx.x * 128;
  const int tid = threadIdx.x;
  const int wid = tid >> 5, lane = tid & 31;
  const int wm = (wid >> 1) * 64;      // wave M offset: 0 / 64
  const int wn = (wid & 1) * 64;       // wave N offset: 0 / 64
  const int fr = lane & 15;
  const int kb = (lane < 16) ? 0 : 8;

  const v8f vzero = {0.f, 0.f, 0.f, 0.f, 0.f, 0.f, 0.f, 0.f};
  v8f acc[4][4];
#pragma unroll
  for (int mt = 0; mt < 4; ++mt)
#pragma unroll
    for (int nt = 0; nt < 4; ++nt) acc[mt][nt] = vzero;

  // Each thread owns one 64-byte row-slab of each tile per K-step.
  const __bf16* Ag = A  + (size_t)(m0 + tid) * K;
  const __bf16* Bg = Bt + (size_t)(n0 + tid) * K;
  __bf16* Al[2] = { &As[0][tid * LD], &As[1][tid * LD] };
  __bf16* Bl[2] = { &Bs[0][tid * LD], &Bs[1][tid * LD] };

#if ASYNC_LDS
  async_cp64(Ag, Al[0]);
  async_cp64(Bg, Bl[0]);
  int buf = 0;
  for (int k0 = 0; k0 < K; k0 += 32) {
    wait_async0();                     // own async copies for `buf` done
    __syncthreads();                   // everyone's copies done / prev compute done
    if (k0 + 32 < K) {                 // prefetch next slab into other buffer
      async_cp64(Ag + k0 + 32, Al[buf ^ 1]);
      async_cp64(Bg + k0 + 32, Bl[buf ^ 1]);
    }
    const __bf16* Ab = As[buf];
    const __bf16* Bb = Bs[buf];
    v16bf bfr[4];
#pragma unroll
    for (int nt = 0; nt < 4; ++nt)
      bfr[nt] = ld_frag16(&Bb[(wn + nt * 16 + fr) * LD + kb]);
#pragma unroll
    for (int mt = 0; mt < 4; ++mt) {
      v16bf af = ld_frag16(&Ab[(wm + mt * 16 + fr) * LD + kb]);
#pragma unroll
      for (int nt = 0; nt < 4; ++nt)
        acc[mt][nt] = wmma_bf16(af, bfr[nt], acc[mt][nt]);
    }
    buf ^= 1;
  }
#else
  for (int k0 = 0; k0 < K; k0 += 32) {
    U128 a0 = *(const U128*)(Ag + k0),      a1 = *(const U128*)(Ag + k0 + 8);
    U128 a2 = *(const U128*)(Ag + k0 + 16), a3 = *(const U128*)(Ag + k0 + 24);
    U128 b0 = *(const U128*)(Bg + k0),      b1 = *(const U128*)(Bg + k0 + 8);
    U128 b2 = *(const U128*)(Bg + k0 + 16), b3 = *(const U128*)(Bg + k0 + 24);
    __syncthreads();
    *(U128*)(Al[0]) = a0; *(U128*)(Al[0] + 8)  = a1;
    *(U128*)(Al[0] + 16) = a2; *(U128*)(Al[0] + 24) = a3;
    *(U128*)(Bl[0]) = b0; *(U128*)(Bl[0] + 8)  = b1;
    *(U128*)(Bl[0] + 16) = b2; *(U128*)(Bl[0] + 24) = b3;
    __syncthreads();
    if (k0 + 32 < K) {
      __builtin_prefetch(Ag + k0 + 32, 0, 0);
      __builtin_prefetch(Bg + k0 + 32, 0, 0);
    }
    v16bf bfr[4];
#pragma unroll
    for (int nt = 0; nt < 4; ++nt)
      bfr[nt] = ld_frag16(&Bs[0][(wn + nt * 16 + fr) * LD + kb]);
#pragma unroll
    for (int mt = 0; mt < 4; ++mt) {
      v16bf af = ld_frag16(&As[0][(wm + mt * 16 + fr) * LD + kb]);
#pragma unroll
      for (int nt = 0; nt < 4; ++nt)
        acc[mt][nt] = wmma_bf16(af, bfr[nt], acc[mt][nt]);
    }
  }
#endif

  const int mofs = (lane >= 16) ? 8 : 0;
#pragma unroll
  for (int nt = 0; nt < 4; ++nt) {
    int n = n0 + wn + nt * 16 + fr;
    float bv = bias ? bias[n] : 0.f;
#pragma unroll
    for (int mt = 0; mt < 4; ++mt) {
#pragma unroll
      for (int r = 0; r < 8; ++r) {
        int m = m0 + wm + mt * 16 + r + mofs;
        float v = acc[mt][nt][r] + bv;
        if (GELU) v = 0.5f * v * (1.f + erff(v * 0.70710678118654752f));
        size_t idx = (size_t)m * N + n;
        if (res)  v += res[idx];
        if (outf) outf[idx] = v;
        if (outb) outb[idx] = (__bf16)v;
      }
    }
  }
}

// ---------------------------------------------------------------------------
// Flash attention with additive bias. Grid (S/64, NH, B), 128 threads/4 waves.
// ---------------------------------------------------------------------------
__global__ void __launch_bounds__(128)
attn_kernel(const __bf16* __restrict__ q, const __bf16* __restrict__ k,
            const __bf16* __restrict__ v, const float* __restrict__ bias,
            __bf16* __restrict__ o) {
  constexpr int LD = 72;               // padded stride for 64-wide tiles
  constexpr float SC = 0.125f;         // D^-0.5
  __shared__ __bf16 Qs[64 * LD];
  __shared__ __bf16 Ks[64 * LD];
  __shared__ __bf16 Vt[64 * LD];       // transposed: [d][t]
  __shared__ __bf16 Ps[64 * LD];

  const int s0   = blockIdx.x * 64;
  const int head = blockIdx.y;
  const int b    = blockIdx.z;
  const int tid = threadIdx.x, wid = tid >> 5, lane = tid & 31;
  const int fr = lane & 15;
  const int kb = (lane < 16) ? 0 : 8;
  const int mofs = (lane >= 16) ? 8 : 0;
  const int sw = wid * 16;
  const int lqr = tid >> 1, lqp = (tid & 1) * 32;   // tile-load row/part

  { // load Q tile [s][d]
    const __bf16* src = q + ((size_t)(b * S_ + s0 + lqr) * H_ + head * D_ + lqp);
    __bf16* dst = &Qs[lqr * LD + lqp];
#if ASYNC_LDS
    async_cp64(src, dst);
#else
    U128 q0 = *(const U128*)src,        q1 = *(const U128*)(src + 8);
    U128 q2 = *(const U128*)(src + 16), q3 = *(const U128*)(src + 24);
    *(U128*)dst = q0; *(U128*)(dst + 8) = q1;
    *(U128*)(dst + 16) = q2; *(U128*)(dst + 24) = q3;
#endif
  }

  const v8f vzero = {0.f, 0.f, 0.f, 0.f, 0.f, 0.f, 0.f, 0.f};
  v8f accO[4]; float mi[8], li[8];
#pragma unroll
  for (int dt = 0; dt < 4; ++dt) accO[dt] = vzero;
#pragma unroll
  for (int r = 0; r < 8; ++r) { mi[r] = -1e30f; li[r] = 0.f; }

  const float* biasb = bias + (((size_t)b * NH_ + head) * S_ + s0) * S_;

  for (int j = 0; j < 8; ++j) {
    const int t0 = j * 64;
    __syncthreads();                   // previous iter done with Ks/Vt/Ps
    { // load K tile [t][d]
      const __bf16* src = k + ((size_t)(b * S_ + t0 + lqr) * H_ + head * D_ + lqp);
      __bf16* dst = &Ks[lqr * LD + lqp];
#if ASYNC_LDS
      async_cp64(src, dst);
#else
      U128 k0 = *(const U128*)src,        k1 = *(const U128*)(src + 8);
      U128 k2 = *(const U128*)(src + 16), k3 = *(const U128*)(src + 24);
      *(U128*)dst = k0; *(U128*)(dst + 8) = k1;
      *(U128*)(dst + 16) = k2; *(U128*)(dst + 24) = k3;
#endif
    }
    { // load V tile transposed into Vt[d][t]
#pragma unroll
      for (int i = 0; i < 16; ++i) {
        int e = tid + i * 128;         // dword index 0..2047
        int t = e >> 5, dc = (e & 31) * 2;
        unsigned int u = *(const unsigned int*)(v + ((size_t)(b * S_ + t0 + t) * H_ +
                                                     head * D_ + dc));
        *(unsigned short*)&Vt[(dc + 0) * LD + t] = (unsigned short)(u & 0xffffu);
        *(unsigned short*)&Vt[(dc + 1) * LD + t] = (unsigned short)(u >> 16);
      }
    }
    wait_async0();
    __syncthreads();

    // scores S = Q K^T  (A = Qs rows, B = Ks rows as N-major)
    v8f sc[4];
#pragma unroll
    for (int nt = 0; nt < 4; ++nt) sc[nt] = vzero;
#pragma unroll
    for (int ks = 0; ks < 64; ks += 32) {
      v16bf af = ld_frag16(&Qs[(sw + fr) * LD + ks + kb]);
#pragma unroll
      for (int nt = 0; nt < 4; ++nt) {
        v16bf bf = ld_frag16(&Ks[(nt * 16 + fr) * LD + ks + kb]);
        sc[nt] = wmma_bf16(af, bf, sc[nt]);
      }
    }

    // scale + bias
    float pv[4][8];
#pragma unroll
    for (int nt = 0; nt < 4; ++nt) {
      int t = t0 + nt * 16 + fr;
#pragma unroll
      for (int r = 0; r < 8; ++r) {
        int srow = sw + r + mofs;
        pv[nt][r] = sc[nt][r] * SC + biasb[(size_t)srow * S_ + t];
      }
    }

    // online softmax per row (rows live in 16-lane halves)
#pragma unroll
    for (int r = 0; r < 8; ++r) {
      float mx = pv[0][r];
#pragma unroll
      for (int nt = 1; nt < 4; ++nt) mx = fmaxf(mx, pv[nt][r]);
#pragma unroll
      for (int off = 8; off > 0; off >>= 1) mx = fmaxf(mx, __shfl_xor(mx, off));
      float mnew = fmaxf(mi[r], mx);
      float alpha = expf(mi[r] - mnew);
      mi[r] = mnew;
      float rsum = 0.f;
#pragma unroll
      for (int nt = 0; nt < 4; ++nt) {
        float e = expf(pv[nt][r] - mnew);
        pv[nt][r] = e; rsum += e;
      }
#pragma unroll
      for (int off = 8; off > 0; off >>= 1) rsum += __shfl_xor(rsum, off);
      li[r] = li[r] * alpha + rsum;
#pragma unroll
      for (int dt = 0; dt < 4; ++dt) accO[dt][r] *= alpha;
    }

    // P -> LDS (bf16) so it can re-enter WMMA as an A fragment
#pragma unroll
    for (int nt = 0; nt < 4; ++nt)
#pragma unroll
      for (int r = 0; r < 8; ++r)
        Ps[(sw + r + mofs) * LD + nt * 16 + fr] = (__bf16)pv[nt][r];
    __syncthreads();

    // O += P * V   (B = Vt rows = columns d, contiguous in t)
#pragma unroll
    for (int ks = 0; ks < 64; ks += 32) {
      v16bf af = ld_frag16(&Ps[(sw + fr) * LD + ks + kb]);
#pragma unroll
      for (int dt = 0; dt < 4; ++dt) {
        v16bf bf = ld_frag16(&Vt[(dt * 16 + fr) * LD + ks + kb]);
        accO[dt] = wmma_bf16(af, bf, accO[dt]);
      }
    }
  }

  // normalize + store
#pragma unroll
  for (int r = 0; r < 8; ++r) {
    float inv = 1.f / li[r];
    int srow = s0 + sw + r + mofs;
#pragma unroll
    for (int dt = 0; dt < 4; ++dt) {
      float ov = accO[dt][r] * inv;
      o[(size_t)(b * S_ + srow) * H_ + head * D_ + dt * 16 + fr] = (__bf16)ov;
    }
  }
}

// ---------------------------------------------------------------------------
extern "C" void kernel_launch(void* const* d_in, const int* in_sizes, int n_in,
                              void* d_out, int out_size, void* d_ws, size_t ws_size,
                              hipStream_t stream) {
  (void)in_sizes; (void)n_in; (void)out_size; (void)ws_size;
  const float* x    = (const float*)d_in[0];
  const float* bias = (const float*)d_in[1];
  const float* ln1g = (const float*)d_in[2];
  const float* ln1b = (const float*)d_in[3];
  const float* wq   = (const float*)d_in[4];
  const float* bq   = (const float*)d_in[5];
  const float* wk   = (const float*)d_in[6];
  const float* bk   = (const float*)d_in[7];
  const float* wv   = (const float*)d_in[8];
  const float* bv   = (const float*)d_in[9];
  const float* wo   = (const float*)d_in[10];
  const float* bo   = (const float*)d_in[11];
  const float* ln2g = (const float*)d_in[12];
  const float* ln2b = (const float*)d_in[13];
  const float* w1   = (const float*)d_in[14];
  const float* b1   = (const float*)d_in[15];
  const float* w2   = (const float*)d_in[16];
  const float* b2   = (const float*)d_in[17];
  const float* flng = (const float*)d_in[18];
  const float* flnb = (const float*)d_in[19];

  char* wp = (char*)d_ws;
  auto take = [&](size_t bytes) -> void* {
    void* r = (void*)wp; wp += (bytes + 255) & ~(size_t)255; return r;
  };
  const size_t nHH = (size_t)H_ * H_;
  const size_t nHF = (size_t)H_ * FF_;
  __bf16* wqT = (__bf16*)take(L_ * nHH * 2);
  __bf16* wkT = (__bf16*)take(L_ * nHH * 2);
  __bf16* wvT = (__bf16*)take(L_ * nHH * 2);
  __bf16* woT = (__bf16*)take(L_ * nHH * 2);
  __bf16* w1T = (__bf16*)take(L_ * nHF * 2);
  __bf16* w2T = (__bf16*)take(L_ * nHF * 2);
  float*  h   = (float*) take((size_t)M_ * H_ * 4);
  __bf16* y   = (__bf16*)take((size_t)M_ * H_ * 2);
  __bf16* qb  = (__bf16*)take((size_t)M_ * H_ * 2);
  __bf16* kb  = (__bf16*)take((size_t)M_ * H_ * 2);
  __bf16* vb  = (__bf16*)take((size_t)M_ * H_ * 2);
  __bf16* ob  = (__bf16*)take((size_t)M_ * H_ * 2);
  __bf16* u   = (__bf16*)take((size_t)M_ * FF_ * 2);

  // weights -> bf16 transposed
  for (int l = 0; l < L_; ++l) {
    wconv_t_kernel<<<(unsigned)((nHH + 255) / 256), 256, 0, stream>>>(wq + l * nHH, wqT + l * nHH, H_, H_);
    wconv_t_kernel<<<(unsigned)((nHH + 255) / 256), 256, 0, stream>>>(wk + l * nHH, wkT + l * nHH, H_, H_);
    wconv_t_kernel<<<(unsigned)((nHH + 255) / 256), 256, 0, stream>>>(wv + l * nHH, wvT + l * nHH, H_, H_);
    wconv_t_kernel<<<(unsigned)((nHH + 255) / 256), 256, 0, stream>>>(wo + l * nHH, woT + l * nHH, H_, H_);
    wconv_t_kernel<<<(unsigned)((nHF + 255) / 256), 256, 0, stream>>>(w1 + l * nHF, w1T + l * nHF, H_, FF_);
    wconv_t_kernel<<<(unsigned)((nHF + 255) / 256), 256, 0, stream>>>(w2 + l * nHF, w2T + l * nHF, FF_, H_);
  }
  copy_f32_kernel<<<(M_ * H_) / 256, 256, 0, stream>>>(x, h, M_ * H_);

  const dim3 gH(H_ / 128, M_ / 128);   // (6, 64)
  const dim3 gF(FF_ / 128, M_ / 128);  // (24, 64)
  const dim3 gA(S_ / 64, NH_, B_);     // (8, 12, 16)

  for (int l = 0; l < L_; ++l) {
    ln_kernel<<<M_, 256, 0, stream>>>(h, ln1g + l * H_, ln1b + l * H_, y, nullptr);
    gemm_bf16_kernel<false><<<gH, 128, 0, stream>>>(y, wqT + l * nHH, bq + l * H_,
                                                    nullptr, nullptr, qb, M_, H_, H_);
    gemm_bf16_kernel<false><<<gH, 128, 0, stream>>>(y, wkT + l * nHH, bk + l * H_,
                                                    nullptr, nullptr, kb, M_, H_, H_);
    gemm_bf16_kernel<false><<<gH, 128, 0, stream>>>(y, wvT + l * nHH, bv + l * H_,
                                                    nullptr, nullptr, vb, M_, H_, H_);
    attn_kernel<<<gA, 128, 0, stream>>>(qb, kb, vb, bias, ob);
    gemm_bf16_kernel<false><<<gH, 128, 0, stream>>>(ob, woT + l * nHH, bo + l * H_,
                                                    h, h, nullptr, M_, H_, H_);
    ln_kernel<<<M_, 256, 0, stream>>>(h, ln2g + l * H_, ln2b + l * H_, y, nullptr);
    gemm_bf16_kernel<true><<<gF, 128, 0, stream>>>(y, w1T + l * nHF, b1 + l * FF_,
                                                   nullptr, nullptr, u, M_, FF_, H_);
    gemm_bf16_kernel<false><<<gH, 128, 0, stream>>>(u, w2T + l * nHF, b2 + l * H_,
                                                    h, h, nullptr, M_, H_, FF_);
  }
  ln_kernel<<<M_, 256, 0, stream>>>(h, flng, flnb, nullptr, (float*)d_out);
}
